// Head_44495861186895
// MI455X (gfx1250) — matrix operand order repaired and checked by reading, not compiled, not run on hardware
//
#include <hip/hip_runtime.h>

// ---- Problem constants (from reference) ----
#define B_BATCH 4
#define T_SEQ   4096
#define E_DIM   1024
#define H_DIM   64
// ALiBi slope = 2^(-8/16) = 2^-0.5 ; score scale = E^-0.5 = 1/32
#define ALIBI_SLOPE 0.70710678118654752f
#define SCORE_SCALE 0.03125f

typedef __attribute__((ext_vector_type(16))) _Float16 v16h;
typedef __attribute__((ext_vector_type(8)))  _Float16 v8h;
typedef __attribute__((ext_vector_type(8)))  float    v8f;
typedef __attribute__((ext_vector_type(4)))  float    v4f;

__device__ __forceinline__ v16h cat8(v8h lo, v8h hi) {
  return __builtin_shufflevector(lo, hi, 0,1,2,3,4,5,6,7,8,9,10,11,12,13,14,15);
}

__device__ __forceinline__ v8f wmma_f16(v16h a, v16h b, v8f c) {
  // D = A(16x32 f16) x B(32x16 f16) + C(16x16 f32)
  return __builtin_amdgcn_wmma_f32_16x16x32_f16(
      false, a, false, b, (short)0, c, false, false);
}

__device__ __forceinline__ v8f vzero8() {
  v8f z = {0.f,0.f,0.f,0.f,0.f,0.f,0.f,0.f};
  return z;
}

// ---- CDNA5 async global->LDS copy (ASYNCcnt-tracked, no VGPR data) ----
__device__ __forceinline__ void async_copy16(unsigned lds_off, const void* gaddr) {
  asm volatile("global_load_async_to_lds_b128 %0, %1, off"
               :: "v"(lds_off), "v"(gaddr) : "memory");
}
#define S_WAIT_ASYNCCNT(n) asm volatile("s_wait_asynccnt " #n ::: "memory")

// LDS byte offset of a __shared__ object = low 32 bits of its generic address
__device__ __forceinline__ unsigned lds_off_of(const void* p) {
  return (unsigned)(uintptr_t)p;
}

// ---- fragment load helpers ----
// A fragment (16x32), contiguous k order used here: lane<16 holds K {0..15}, lane>=16 holds
// K {16..31}; two contiguous 8-half chunks at +hi*16 and +hi*16+8.
__device__ __forceinline__ v16h loadA_f16(const _Float16* row, int hi) {
  return cat8(*(const v8h*)(row + hi * 16), *(const v8h*)(row + hi * 16 + 8));
}
// Same A layout, source f32: load 8+8 floats, convert in registers.
__device__ __forceinline__ v16h loadA_f32(const float* xrow, int k0, int hi) {
  const float* pa = xrow + k0 + hi * 16;
  v4f f0 = *(const v4f*)(pa);
  v4f f1 = *(const v4f*)(pa + 4);
  const float* pb = xrow + k0 + hi * 16 + 8;
  v4f f2 = *(const v4f*)(pb);
  v4f f3 = *(const v4f*)(pb + 4);
  v16h a;
  #pragma unroll
  for (int i = 0; i < 4; ++i) {
    a[i]      = (_Float16)f0[i];
    a[4 + i]  = (_Float16)f1[i];
    a[8 + i]  = (_Float16)f2[i];
    a[12 + i] = (_Float16)f3[i];
  }
  return a;
}
// B fragment (32x16), contiguous k order used here: lane n holds 16 consecutive K values of column n.
__device__ __forceinline__ v16h loadB_f16(const _Float16* colbase) {
  return cat8(*(const v8h*)(colbase), *(const v8h*)(colbase + 8));
}

// -------------------------------------------------------------------
// Kernel 0: transpose + convert Wq/Wk/Wv [E,H] f32 -> WT [3][H][E] f16
// -------------------------------------------------------------------
__global__ void wt_kernel(const float* __restrict__ Wq,
                          const float* __restrict__ Wk,
                          const float* __restrict__ Wv,
                          _Float16* __restrict__ WT) {
  int idx  = blockIdx.x * blockDim.x + threadIdx.x;   // 3*64*1024 exact
  int proj = idx >> 16;
  int rem  = idx & 65535;
  int h    = rem >> 10;
  int e    = rem & 1023;
  const float* W = (proj == 0) ? Wq : (proj == 1) ? Wk : Wv;
  WT[(size_t)proj * H_DIM * E_DIM + (size_t)h * E_DIM + e] =
      (_Float16)W[(size_t)e * H_DIM + h];
}

// -------------------------------------------------------------------
// Kernel 1: projections. Two-stage (ping-pong) software pipeline:
// no register copies between stages, loads stay one stage ahead.
//   proj 0 -> qh [B][T][H]   proj 1 -> kh [B][T][H]   proj 2 -> vhT [B][H][T]
// -------------------------------------------------------------------
__global__ __launch_bounds__(32)
void proj_kernel(const float* __restrict__ q,
                 const float* __restrict__ k,
                 const float* __restrict__ v,
                 const _Float16* __restrict__ WT,
                 _Float16* __restrict__ qh,
                 _Float16* __restrict__ kh,
                 _Float16* __restrict__ vhT) {
  const int proj = blockIdx.y;
  const float* X = (proj == 0) ? q : (proj == 1) ? k : v;
  const _Float16* W = WT + (size_t)proj * H_DIM * E_DIM;

  const int tile = blockIdx.x;
  const int b    = tile >> 8;
  const int r0   = (tile & 255) << 4;
  const int lane = threadIdx.x & 31;
  const int n    = lane & 15;
  const int hi   = lane >> 4;

  const float* xrow = X + ((size_t)b * T_SEQ + r0 + n) * E_DIM;
  const _Float16* wcol = W + (size_t)n * E_DIM + hi * 16;  // + nt*16*E + k0

  v8f acc[4] = {vzero8(), vzero8(), vzero8(), vzero8()};

  // stage A: chunk k0 = 0
  v16h aA = loadA_f32(xrow, 0, hi);
  v16h bA[4];
  #pragma unroll
  for (int nt = 0; nt < 4; ++nt)
    bA[nt] = loadB_f16(wcol + (size_t)nt * 16 * E_DIM);

  // ping-pong: iteration consumes chunks (k0-32, k0) and preloads k0+32
  for (int k0 = 32; k0 < E_DIM - 32; k0 += 64) {
    v16h aB = loadA_f32(xrow, k0, hi);
    v16h bB[4];
    #pragma unroll
    for (int nt = 0; nt < 4; ++nt)
      bB[nt] = loadB_f16(wcol + (size_t)nt * 16 * E_DIM + k0);

    #pragma unroll
    for (int nt = 0; nt < 4; ++nt) acc[nt] = wmma_f16(aA, bA[nt], acc[nt]);

    aA = loadA_f32(xrow, k0 + 32, hi);
    #pragma unroll
    for (int nt = 0; nt < 4; ++nt)
      bA[nt] = loadB_f16(wcol + (size_t)nt * 16 * E_DIM + k0 + 32);

    #pragma unroll
    for (int nt = 0; nt < 4; ++nt) acc[nt] = wmma_f16(aB, bB[nt], acc[nt]);
  }
  // epilogue: final chunk E_DIM-32 into stage B, then drain both stages
  {
    const int k0 = E_DIM - 32;
    v16h aB = loadA_f32(xrow, k0, hi);
    v16h bB[4];
    #pragma unroll
    for (int nt = 0; nt < 4; ++nt)
      bB[nt] = loadB_f16(wcol + (size_t)nt * 16 * E_DIM + k0);
    #pragma unroll
    for (int nt = 0; nt < 4; ++nt) acc[nt] = wmma_f16(aA, bA[nt], acc[nt]);
    #pragma unroll
    for (int nt = 0; nt < 4; ++nt) acc[nt] = wmma_f16(aB, bB[nt], acc[nt]);
  }

  // branch-free epilogue: select (dst, strides) once per block
  _Float16* dst;
  size_t rowStride, colStride;
  if (proj == 0) {
    dst = qh + (size_t)b * T_SEQ * H_DIM;  rowStride = H_DIM;  colStride = 1;
  } else if (proj == 1) {
    dst = kh + (size_t)b * T_SEQ * H_DIM;  rowStride = H_DIM;  colStride = 1;
  } else {
    dst = vhT + (size_t)b * H_DIM * T_SEQ; rowStride = 1;      colStride = T_SEQ;
  }
  #pragma unroll
  for (int nt = 0; nt < 4; ++nt) {
    #pragma unroll
    for (int j = 0; j < 8; ++j) {
      const size_t row = r0 + j + 8 * hi;
      const size_t col = nt * 16 + n;
      dst[row * rowStride + col * colStride] = (_Float16)acc[nt][j];
    }
  }
}

// -------------------------------------------------------------------
// Kernel 2: causal flash attention with ALiBi.
// K/V tiles double-buffered in LDS via global_load_async_to_lds_b128;
// tile kt+1 prefetched while computing tile kt (ASYNCcnt pipelining).
// -------------------------------------------------------------------
__global__ __launch_bounds__(32)
void attn_kernel(const _Float16* __restrict__ qh,
                 const _Float16* __restrict__ kh,
                 const _Float16* __restrict__ vhT,
                 float* __restrict__ out) {
  __shared__ __align__(16) _Float16 Kbuf[2][32 * 64];  // [key-local][h]
  __shared__ __align__(16) _Float16 Vbuf[2][64 * 32];  // [h][t-local]
  __shared__ __align__(16) _Float16 Plds[16 * 32];     // [row][key-local]

  const int tile = blockIdx.x;
  const int b    = tile >> 8;
  const int q0   = (tile & 255) << 4;
  const int lane = threadIdx.x & 31;
  const int n    = lane & 15;
  const int hi   = lane >> 4;

  // A fragments of qh rows q0..q0+15 (K = H = 64, two K=32 tiles)
  const _Float16* qrow = qh + ((size_t)b * T_SEQ + q0 + n) * H_DIM;
  const v16h a0 = loadA_f16(qrow, hi);
  const v16h a1 = loadA_f16(qrow + 32, hi);

  v8f acc[4] = {vzero8(), vzero8(), vzero8(), vzero8()};
  float mrow[8], lrow[8];
  #pragma unroll
  for (int j = 0; j < 8; ++j) { mrow[j] = -INFINITY; lrow[j] = 0.f; }

  const int kt_cnt = (q0 + 47) >> 5;   // key tiles with j0 <= q0+15

  // ---- async prefetch of one 32-key K/V tile: 16 b128 instructions ----
  auto prefetch_tile = [&](int kt, int buf) {
    const int j0 = kt * 32;
    // K tile: 32 rows x 128B; lane copies its row in 8 x 16B chunks
    const char* kg =
        (const char*)(kh + ((size_t)b * T_SEQ + j0 + lane) * H_DIM);
    const unsigned klds = lds_off_of(&Kbuf[buf][lane * 64]);
    #pragma unroll
    for (int c = 0; c < 8; ++c) async_copy16(klds + c * 16, kg + c * 16);
    // V tile: 64 rows x 64B; lane covers rows {lane, lane+32}, 4 chunks each
    #pragma unroll
    for (int c = 0; c < 8; ++c) {
      const int row = lane + 32 * (c >> 2);
      const char* vg =
          (const char*)(vhT + ((size_t)b * H_DIM + row) * T_SEQ + j0) +
          (c & 3) * 16;
      const unsigned vlds = lds_off_of(&Vbuf[buf][row * 32]) + (c & 3) * 16;
      async_copy16(vlds, vg);
    }
  };

  prefetch_tile(0, 0);

  for (int kt = 0; kt < kt_cnt; ++kt) {
    const int j0  = kt * 32;
    const int buf = kt & 1;

    if (kt + 1 < kt_cnt) {
      prefetch_tile(kt + 1, (kt + 1) & 1);  // ASYNCcnt -> 32
      S_WAIT_ASYNCCNT(0x10);                // tile kt's 16 loads retired
    } else {
      S_WAIT_ASYNCCNT(0x0);
    }

    // ---- S = qh . kh^T for 32 keys (two 16x16 score tiles) ----
    const _Float16* kr0 = &Kbuf[buf][(size_t)n * 64];         // keys 0..15
    const _Float16* kr1 = &Kbuf[buf][(size_t)(16 + n) * 64];  // keys 16..31
    const v16h b00 = loadB_f16(kr0 + hi * 16);
    const v16h b01 = loadB_f16(kr0 + 32 + hi * 16);
    const v16h b10 = loadB_f16(kr1 + hi * 16);
    const v16h b11 = loadB_f16(kr1 + 32 + hi * 16);

    v8f s0 = wmma_f16(a0, b00, vzero8());
    s0     = wmma_f16(a1, b01, s0);
    v8f s1 = wmma_f16(a0, b10, vzero8());
    s1     = wmma_f16(a1, b11, s1);

    // ---- scale + ALiBi + causal mask + online softmax ----
    #pragma unroll
    for (int j = 0; j < 8; ++j) {
      const int irow = q0 + j + 8 * hi;
      const int jc0  = j0 + n;
      const int jc1  = j0 + 16 + n;
      float e0 = s0[j] * SCORE_SCALE + ALIBI_SLOPE * (float)(jc0 - irow);
      float e1 = s1[j] * SCORE_SCALE + ALIBI_SLOPE * (float)(jc1 - irow);
      if (jc0 > irow) e0 = -INFINITY;
      if (jc1 > irow) e1 = -INFINITY;

      float rmax = fmaxf(e0, e1);
      rmax = fmaxf(rmax, __shfl_xor(rmax, 1));
      rmax = fmaxf(rmax, __shfl_xor(rmax, 2));
      rmax = fmaxf(rmax, __shfl_xor(rmax, 4));
      rmax = fmaxf(rmax, __shfl_xor(rmax, 8));

      const float mnew = fmaxf(mrow[j], rmax);
      const float corr = __expf(mrow[j] - mnew);
      const float p0   = __expf(e0 - mnew);
      const float p1   = __expf(e1 - mnew);

      float ps = p0 + p1;
      ps += __shfl_xor(ps, 1);
      ps += __shfl_xor(ps, 2);
      ps += __shfl_xor(ps, 4);
      ps += __shfl_xor(ps, 8);

      lrow[j] = lrow[j] * corr + ps;
      mrow[j] = mnew;
      #pragma unroll
      for (int nt = 0; nt < 4; ++nt) acc[nt][j] *= corr;

      Plds[(j + 8 * hi) * 32 + n]      = (_Float16)p0;
      Plds[(j + 8 * hi) * 32 + 16 + n] = (_Float16)p1;
    }
    __syncthreads();

    // ---- PV: acc += P (16x32) . V-tile (32x16) per 16-wide h tile ----
    const _Float16* prow = Plds + n * 32;
    const v16h pa = cat8(*(const v8h*)(prow + hi * 16),
                         *(const v8h*)(prow + hi * 16 + 8));
    #pragma unroll
    for (int nt = 0; nt < 4; ++nt) {
      const _Float16* vp = &Vbuf[buf][(size_t)(nt * 16 + n) * 32 + hi * 16];
      const v16h bv = loadB_f16(vp);
      acc[nt] = wmma_f16(pa, bv, acc[nt]);
    }
    __syncthreads();
  }

  // ---- normalize and write f32 output ----
  #pragma unroll
  for (int j = 0; j < 8; ++j) {
    const float inv = 1.0f / lrow[j];
    const int row = q0 + j + 8 * hi;
    #pragma unroll
    for (int nt = 0; nt < 4; ++nt) {
      out[((size_t)b * T_SEQ + row) * H_DIM + nt * 16 + n] = acc[nt][j] * inv;
    }
  }
}

// -------------------------------------------------------------------
extern "C" void kernel_launch(void* const* d_in, const int* in_sizes, int n_in,
                              void* d_out, int out_size, void* d_ws, size_t ws_size,
                              hipStream_t stream) {
  const float* q  = (const float*)d_in[0];
  const float* k  = (const float*)d_in[1];
  const float* v  = (const float*)d_in[2];
  const float* Wq = (const float*)d_in[3];
  const float* Wk = (const float*)d_in[4];
  const float* Wv = (const float*)d_in[5];
  float* out = (float*)d_out;

  char* ws = (char*)d_ws;
  _Float16* WT  = (_Float16*)(ws);                               // 384 KB
  _Float16* qh  = (_Float16*)(ws + (512u << 10));                // 2 MB
  _Float16* kh  = (_Float16*)(ws + (512u << 10) + (2u << 20));   // 2 MB
  _Float16* vhT = (_Float16*)(ws + (512u << 10) + (4u << 20));   // 2 MB

  wt_kernel<<<768, 256, 0, stream>>>(Wq, Wk, Wv, WT);

  dim3 gproj(B_BATCH * T_SEQ / 16, 3);
  proj_kernel<<<gproj, 32, 0, stream>>>(q, k, v, WT, qh, kh, vhT);

  attn_kernel<<<B_BATCH * T_SEQ / 16, 32, 0, stream>>>(qh, kh, vhT, out);
}
